// SelfAttention_27161373180656
// MI455X (gfx1250) — matrix operand order repaired and checked by reading, not compiled, not run on hardware
//
#include <hip/hip_runtime.h>

typedef __bf16 bf16_t;
typedef __attribute__((ext_vector_type(16))) __bf16 v16bf;
typedef __attribute__((ext_vector_type(8)))  __bf16 v8bf;
typedef __attribute__((ext_vector_type(8)))  float  v8f;

// A-matrix 16x32 bf16 fragment: lane half kh holds k = base + {0..15} (contiguous)
__device__ __forceinline__ v16bf load_a_frag(const bf16_t* base /*16B aligned*/) {
    v8bf lo = *(const v8bf*)(base);
    v8bf hi = *(const v8bf*)(base + 8);
    v16bf a;
#pragma unroll
    for (int j = 0; j < 8; j++) { a[j] = lo[j]; a[j + 8] = hi[j]; }
    return a;
}

__device__ __forceinline__ v8f wmma_bf16(v16bf a, v16bf b, v8f c) {
    return __builtin_amdgcn_wmma_f32_16x16x32_bf16(false, a, false, b, (short)0, c, false, false);
}

// ---------------------------------------------------------------------------
// Kernel 0: convert weights f32 -> bf16, fold q scaling (C^-0.5) into Wq.
// Wpack layout: [Wq_scaled | Wk | Wv | Wg | Wo], each 256*256 bf16, row-major [e][d].
// ---------------------------------------------------------------------------
__global__ void __launch_bounds__(256) convert_kernel(
    const float* __restrict__ Wq, const float* __restrict__ Wk,
    const float* __restrict__ Wv, const float* __restrict__ Wg,
    const float* __restrict__ Wo, bf16_t* __restrict__ Wpack) {
    int i = blockIdx.x * 256 + threadIdx.x;
    if (i >= 65536) return;
    Wpack[i]          = (bf16_t)(Wq[i] * 0.17677669529663687f);  // 1/sqrt(32)
    Wpack[65536 + i]  = (bf16_t)Wk[i];
    Wpack[131072 + i] = (bf16_t)Wv[i];
    Wpack[196608 + i] = (bf16_t)Wg[i];
    Wpack[262144 + i] = (bf16_t)Wo[i];
}

// ---------------------------------------------------------------------------
// Kernel 1: projections. grid = (128 batches, 4 matrices), 256 thr (8 waves).
// LDS: X_b bf16 (128KB) + W_sel bf16 (128KB) = 256KB of the WGP's 320KB.
// Each wave: 4 macro-tiles of 64x32 (4x2 register blocking, 8 acc per wave),
// 8 WMMAs per K-step from 6 fragment loads (0.75 loads/WMMA).
// ---------------------------------------------------------------------------
__global__ void __launch_bounds__(256) proj_kernel(
    const float* __restrict__ X, const float* __restrict__ gbias,
    const bf16_t* __restrict__ Wpack,
    bf16_t* __restrict__ Qw, bf16_t* __restrict__ Kw,
    bf16_t* __restrict__ Vw, bf16_t* __restrict__ Gw) {
    extern __shared__ char smem[];
    bf16_t* Xs = (bf16_t*)smem;                       // [l][d] bf16, 128KB
    bf16_t* Ws = (bf16_t*)(smem + 131072);            // [e][d] bf16, 128KB
    const int tid = threadIdx.x, wave = tid >> 5, lane = tid & 31;
    const int b = blockIdx.x, sel = blockIdx.y;       // sel: 0=q 1=k 2=v 3=g

    const float4* Xg = (const float4*)(X + (size_t)b * 65536);
    for (int i = tid; i < 16384; i += 256) {
        float4 x = Xg[i];
        bf16_t* p = Xs + i * 4;
        p[0] = (bf16_t)x.x; p[1] = (bf16_t)x.y; p[2] = (bf16_t)x.z; p[3] = (bf16_t)x.w;
    }
    const uint4* Wg4 = (const uint4*)(Wpack + (size_t)sel * 65536);
    uint4* Ws4 = (uint4*)Ws;
    for (int i = tid; i < 8192; i += 256) Ws4[i] = Wg4[i];
    __syncthreads();

    const int lane15 = lane & 15, kh = lane >> 4;
    const v8f vzero = {};

    for (int mt = wave; mt < 32; mt += 8) {           // 4x8 grid of 64x32 macro-tiles
        const int rm = (mt >> 3) * 64, cn = (mt & 7) * 32;
        v8f acc[4][2];
#pragma unroll
        for (int i = 0; i < 4; i++)
#pragma unroll
            for (int j = 0; j < 2; j++) acc[i][j] = vzero;
        const bf16_t* abase = Xs + (rm + lane15) * 256 + kh * 16;
        const bf16_t* bbase = Ws + (cn + lane15) * 256 + kh * 16;
#pragma unroll
        for (int kk = 0; kk < 8; kk++) {
            v16bf a0 = load_a_frag(abase + kk * 32);
            v16bf a1 = load_a_frag(abase + 16 * 256 + kk * 32);
            v16bf a2 = load_a_frag(abase + 32 * 256 + kk * 32);
            v16bf a3 = load_a_frag(abase + 48 * 256 + kk * 32);
            v16bf b0 = *(const v16bf*)(bbase + kk * 32);
            v16bf b1 = *(const v16bf*)(bbase + 16 * 256 + kk * 32);
            acc[0][0] = wmma_bf16(a0, b0, acc[0][0]);
            acc[0][1] = wmma_bf16(a0, b1, acc[0][1]);
            acc[1][0] = wmma_bf16(a1, b0, acc[1][0]);
            acc[1][1] = wmma_bf16(a1, b1, acc[1][1]);
            acc[2][0] = wmma_bf16(a2, b0, acc[2][0]);
            acc[2][1] = wmma_bf16(a2, b1, acc[2][1]);
            acc[3][0] = wmma_bf16(a3, b0, acc[3][0]);
            acc[3][1] = wmma_bf16(a3, b1, acc[3][1]);
        }
        const int n = lane15, mh = kh;                // C/D: m = r + 8*mh, n = lane&15
        if (sel == 3) {
#pragma unroll
            for (int i = 0; i < 4; i++)
#pragma unroll
                for (int j = 0; j < 2; j++)
#pragma unroll
                    for (int r = 0; r < 8; r++) {
                        int m = rm + i * 16 + mh * 8 + r;
                        int e = cn + j * 16 + n;
                        float g = 1.0f / (1.0f + __expf(-(acc[i][j][r] + gbias[e])));
                        Gw[((size_t)b * 256 + m) * 256 + e] = (bf16_t)g;
                    }
        } else {
            bf16_t* dst = (sel == 0) ? Qw : (sel == 1 ? Kw : Vw);
#pragma unroll
            for (int i = 0; i < 4; i++)
#pragma unroll
                for (int j = 0; j < 2; j++)
#pragma unroll
                    for (int r = 0; r < 8; r++) {
                        int m = rm + i * 16 + mh * 8 + r;
                        int e = cn + j * 16 + n;
                        int h = e >> 5, c = e & 31;
                        dst[(((size_t)b * 8 + h) * 256 + m) * 32 + c] = (bf16_t)acc[i][j][r];
                    }
        }
    }
}

// ---------------------------------------------------------------------------
// Kernel 2: attention per (b,h). grid = 1024, 256 thr (8 waves).
// K staged via GLOBAL_LOAD_ASYNC_TO_LDS_B128 (ASYNCcnt DMA path); V staged
// transposed. logits: 1 WMMA per key tile (K=c=32, Qa held in regs).
// Softmax with 2 lanes/row; probs unnormalized bf16; 1/sum folded into wavg.
// wavg: A-fragment reused across both N-tiles.
// ---------------------------------------------------------------------------
__global__ void __launch_bounds__(256) attn_kernel(
    const float* __restrict__ nb_bias,                  // [8][256][256]
    const bf16_t* __restrict__ Qw, const bf16_t* __restrict__ Kw,
    const bf16_t* __restrict__ Vw, bf16_t* __restrict__ wavg /*[b][l][hc]*/) {
    extern __shared__ char smem[];
    float*  lg    = (float*)smem;                        // 131072 B
    bf16_t* Ks    = (bf16_t*)(smem + 131072);            // 16KB  [key][c]
    bf16_t* Vt    = (bf16_t*)(smem + 147456);            // 16KB  [c][key]
    bf16_t* probs = (bf16_t*)(smem + 163840);            // 64KB
    float*  invs  = (float*)(smem + 229376);             // 512 B

    const int tid = threadIdx.x, wave = tid >> 5, lane = tid & 31;
    const int bh = blockIdx.x, h = bh & 7, b = bh >> 3;

    const bf16_t* Kh = Kw + (size_t)bh * 8192;
    const bf16_t* Vh = Vw + (size_t)bh * 8192;

    // K: 16KB direct-copy via async DMA to LDS (no static LDS -> dynamic base = 0)
    for (int i = tid; i < 1024; i += 256) {
        unsigned dst = 131072u + (unsigned)i * 16u;      // LDS byte offset of Ks[i*8]
        const char* src = (const char*)Kh + (size_t)i * 16;
        asm volatile("global_load_async_to_lds_b128 %0, %1, off"
                     :: "v"(dst), "v"(src) : "memory");
    }
    // V: transpose into LDS [c][key]
    for (int i = tid; i < 8192; i += 256) {
        int kk = i >> 5, c = i & 31;
        Vt[c * 256 + kk] = Vh[i];
    }
    asm volatile("s_wait_asynccnt 0x0" ::: "memory");
    __syncthreads();

    const int lane15 = lane & 15, kh = lane >> 4;
    float*  lg_w  = lg + wave * 16 * 256;
    bf16_t* pr_w  = probs + wave * 16 * 256;
    float*  inv_w = invs + wave * 16;
    const v8f vzero = {};

    for (int qt = wave; qt < 16; qt += 8) {
        const bf16_t* qrow = Qw + (size_t)bh * 8192 + (qt * 16 + lane15) * 32 + kh * 16;
        v16bf qa = load_a_frag(qrow);                    // K=c=32: single fragment
        const float* biasq = nb_bias + ((size_t)h * 256 + qt * 16) * 256;
        for (int nt = 0; nt < 16; nt++) {
            v16bf bb = *(const v16bf*)(Ks + (nt * 16 + lane15) * 32 + kh * 16);
            v8f acc = wmma_bf16(qa, bb, vzero);
            int n = lane15, mh = kh;
#pragma unroll
            for (int r = 0; r < 8; r++) {
                int m = mh * 8 + r;
                lg_w[m * 256 + nt * 16 + n] = acc[r] + biasq[(size_t)m * 256 + nt * 16 + n];
            }
        }
        // softmax over 256 keys: lanes (m, m+16) split row m in halves
        {
            int lr = lane15, half = kh;
            const float* rowp = lg_w + lr * 256 + half * 128;
            float mx = -3.0e38f;
            for (int i = 0; i < 128; i++) mx = fmaxf(mx, rowp[i]);
            mx = fmaxf(mx, __shfl_xor(mx, 16, 32));
            bf16_t* prow = pr_w + lr * 256 + half * 128;
            float s = 0.f;
            for (int i = 0; i < 128; i++) {
                float e = __expf(rowp[i] - mx);
                prow[i] = (bf16_t)e;                     // unnormalized
                s += e;
            }
            s += __shfl_xor(s, 16, 32);
            if (half == 0) inv_w[lr] = 1.0f / s;
        }
        // wavg = probs(16x256) @ V(256x32): A reused across both N-tiles
        {
            v8f acc0 = vzero, acc1 = vzero;
            const bf16_t* arow = pr_w + lane15 * 256 + kh * 16;
            const bf16_t* brow = Vt + lane15 * 256 + kh * 16;
#pragma unroll
            for (int kk = 0; kk < 8; kk++) {
                v16bf a  = load_a_frag(arow + kk * 32);
                v16bf b0 = *(const v16bf*)(brow + kk * 32);
                v16bf b1 = *(const v16bf*)(brow + 16 * 256 + kk * 32);
                acc0 = wmma_bf16(a, b0, acc0);
                acc1 = wmma_bf16(a, b1, acc1);
            }
            int n = lane15, mh = kh;
#pragma unroll
            for (int r = 0; r < 8; r++) {
                int m = mh * 8 + r;
                int q = qt * 16 + m;
                float inv = inv_w[m];
                bf16_t* orow = wavg + ((size_t)b * 256 + q) * 256 + h * 32 + n;
                orow[0]  = (bf16_t)(acc0[r] * inv);
                orow[16] = (bf16_t)(acc1[r] * inv);
            }
        }
    }
}

// ---------------------------------------------------------------------------
// Kernel 3: out = (gate .* wavg) @ Wo^T + bo.  grid = 128, 256 thr.
// LDS: gated (128KB) + Wo (128KB); 4x2 register blocking like proj.
// ---------------------------------------------------------------------------
__global__ void __launch_bounds__(256) out_kernel(
    const float* __restrict__ bo, const bf16_t* __restrict__ Wo,
    const bf16_t* __restrict__ gate, const bf16_t* __restrict__ wavg,
    float* __restrict__ out) {
    extern __shared__ char smem[];
    bf16_t* Gs  = (bf16_t*)smem;                         // gated [l][e] bf16, 128KB
    bf16_t* Wos = (bf16_t*)(smem + 131072);              // Wo [o][e] bf16, 128KB
    const int tid = threadIdx.x, wave = tid >> 5, lane = tid & 31;
    const int b = blockIdx.x;

    const uint32_t* gp = (const uint32_t*)(gate + (size_t)b * 65536);
    const uint32_t* wp = (const uint32_t*)(wavg + (size_t)b * 65536);
    uint32_t* gsp = (uint32_t*)Gs;
    for (int i = tid; i < 32768; i += 256) {
        union { uint32_t u; __bf16 h[2]; } a, bb, c;
        a.u = gp[i]; bb.u = wp[i];
        c.h[0] = (bf16_t)((float)a.h[0] * (float)bb.h[0]);
        c.h[1] = (bf16_t)((float)a.h[1] * (float)bb.h[1]);
        gsp[i] = c.u;
    }
    const uint4* Wo4 = (const uint4*)Wo;
    uint4* Ws4 = (uint4*)Wos;
    for (int i = tid; i < 8192; i += 256) Ws4[i] = Wo4[i];
    __syncthreads();

    const int lane15 = lane & 15, kh = lane >> 4;
    const v8f vzero = {};
    for (int mt = wave; mt < 32; mt += 8) {              // 64x32 macro-tiles
        const int rm = (mt >> 3) * 64, cn = (mt & 7) * 32;
        v8f acc[4][2];
#pragma unroll
        for (int i = 0; i < 4; i++)
#pragma unroll
            for (int j = 0; j < 2; j++) acc[i][j] = vzero;
        const bf16_t* abase = Gs + (rm + lane15) * 256 + kh * 16;
        const bf16_t* bbase = Wos + (cn + lane15) * 256 + kh * 16;
#pragma unroll
        for (int kk = 0; kk < 8; kk++) {
            v16bf a0 = load_a_frag(abase + kk * 32);
            v16bf a1 = load_a_frag(abase + 16 * 256 + kk * 32);
            v16bf a2 = load_a_frag(abase + 32 * 256 + kk * 32);
            v16bf a3 = load_a_frag(abase + 48 * 256 + kk * 32);
            v16bf b0 = *(const v16bf*)(bbase + kk * 32);
            v16bf b1 = *(const v16bf*)(bbase + 16 * 256 + kk * 32);
            acc[0][0] = wmma_bf16(a0, b0, acc[0][0]);
            acc[0][1] = wmma_bf16(a0, b1, acc[0][1]);
            acc[1][0] = wmma_bf16(a1, b0, acc[1][0]);
            acc[1][1] = wmma_bf16(a1, b1, acc[1][1]);
            acc[2][0] = wmma_bf16(a2, b0, acc[2][0]);
            acc[2][1] = wmma_bf16(a2, b1, acc[2][1]);
            acc[3][0] = wmma_bf16(a3, b0, acc[3][0]);
            acc[3][1] = wmma_bf16(a3, b1, acc[3][1]);
        }
        const int n = lane15, mh = kh;
#pragma unroll
        for (int j = 0; j < 2; j++) {
            float bias = bo[cn + j * 16 + n];
#pragma unroll
            for (int i = 0; i < 4; i++)
#pragma unroll
                for (int r = 0; r < 8; r++) {
                    int m = rm + i * 16 + mh * 8 + r;
                    out[((size_t)b * 256 + m) * 256 + cn + j * 16 + n] = acc[i][j][r] + bias;
                }
        }
    }
}

// ---------------------------------------------------------------------------
extern "C" void kernel_launch(void* const* d_in, const int* in_sizes, int n_in,
                              void* d_out, int out_size, void* d_ws, size_t ws_size,
                              hipStream_t stream) {
    (void)in_sizes; (void)n_in; (void)out_size; (void)ws_size;
    const float* X   = (const float*)d_in[0];   // [1,128,256,256]
    const float* nbb = (const float*)d_in[1];   // [1,8,256,256]
    const float* Wq  = (const float*)d_in[2];
    const float* Wk  = (const float*)d_in[3];
    const float* Wv  = (const float*)d_in[4];
    const float* Wg  = (const float*)d_in[5];
    const float* gb  = (const float*)d_in[6];
    const float* Wo  = (const float*)d_in[7];
    const float* bo  = (const float*)d_in[8];
    float* out = (float*)d_out;

    char* ws = (char*)d_ws;
    bf16_t* Wpack = (bf16_t*)ws;                               // 640 KB
    size_t off = (size_t)5 * 65536 * 2;
    off = (off + 255) & ~(size_t)255;
    bf16_t* Qw  = (bf16_t*)(ws + off); off += (size_t)128 * 8 * 256 * 32 * 2;  // 16MB
    bf16_t* Kw  = (bf16_t*)(ws + off); off += (size_t)128 * 8 * 256 * 32 * 2;  // 16MB
    bf16_t* Vw  = (bf16_t*)(ws + off); off += (size_t)128 * 8 * 256 * 32 * 2;  // 16MB
    bf16_t* Gw  = (bf16_t*)(ws + off); off += (size_t)128 * 256 * 256 * 2;     // 16MB
    bf16_t* AVw = (bf16_t*)(ws + off);                                         // 16MB

    convert_kernel<<<256, 256, 0, stream>>>(Wq, Wk, Wv, Wg, Wo, Wpack);
    dim3 gp(128, 4);
    proj_kernel<<<gp, 256, 262144, stream>>>(X, gb, Wpack, Qw, Kw, Vw, Gw);
    attn_kernel<<<1024, 256, 229888, stream>>>(nbb, Qw, Kw, Vw, AVw);
    out_kernel<<<128, 256, 262144, stream>>>(bo, Wpack + 4 * 65536, Gw, AVw, out);
}
